// AttnAggregator_65816078844625
// MI455X (gfx1250) — matrix operand order repaired and checked by reading, not compiled, not run on hardware
//
#include <hip/hip_runtime.h>
#include <hip/hip_bf16.h>

// ---------- types ----------
typedef __attribute__((ext_vector_type(16))) __bf16 v16bf;
typedef __attribute__((ext_vector_type(8)))  float  v8f;
typedef __attribute__((ext_vector_type(4)))  float  f32x4;

#define DIN   256
#define DOUT  64
#define NT_N  4      // DOUT / 16 output column tiles
#define KSTEPS 8     // DIN / 32 K steps

union ABf { v16bf v; __bf16 e[16]; };
union CF  { v8f   v; float  e[8];  };

// ---------------------------------------------------------------------------
// Kernel 0: convert W_proj [64,256] f32 -> bf16, pre-swizzled into the exact
// B-matrix (32x16 K-major) WMMA fragment layout:
//   fragment f = (ntile, kstep, lane); lane%16 = output col, lane/16 = K-half.
// Each fragment is 16 contiguous bf16 (32B) -> one aligned v16bf load later.
// ---------------------------------------------------------------------------
__global__ void prep_w_kernel(const float* __restrict__ W,
                              unsigned short* __restrict__ Wbf_raw) {
  __bf16* Wbf = (__bf16*)Wbf_raw;
  int f = blockIdx.x * blockDim.x + threadIdx.x;   // 4*8*32 = 1024 fragments
  if (f >= NT_N * KSTEPS * 32) return;
  int ntile = f >> 8;
  int rem   = f & 255;
  int kstep = rem >> 5;
  int lane  = rem & 31;
  int o     = ntile * 16 + (lane & 15);            // output column (row of W)
  int kbase = kstep * 32 + (lane >> 4) * 8;        // K start for this lane
  __bf16* dst = Wbf + (size_t)f * 16;
#pragma unroll
  for (int j = 0; j < 16; ++j) dst[j] = (__bf16)W[o * DIN + kbase + (j < 8 ? j : j + 8)];
}

// ---------------------------------------------------------------------------
// Kernel 1: P[N,64] = embed[N,256] @ W^T via v_wmma_f32_16x16x32_bf16.
// One wave per 16-row tile; 8 K-steps x 4 N-tiles = 32 WMMAs per wave.
// A-fragment layout (16-bit A 16x32): lane<16 holds K {k0..k0+7, k0+16..k0+23},
// lane>=16 holds K {k0+8..k0+15, k0+24..k0+31}, packed as consecutive-K pairs.
// ---------------------------------------------------------------------------
__global__ void proj_kernel(const float* __restrict__ E,
                            const unsigned short* __restrict__ Wbf_raw,
                            float* __restrict__ P, int nTiles) {
  const v16bf* Wf = (const v16bf*)Wbf_raw;
  int wave = threadIdx.x >> 5;
  int lane = threadIdx.x & 31;
  int tile = blockIdx.x * 8 + wave;
  if (tile >= nTiles) return;                       // wave-uniform

  int m     = lane & 15;
  int khalf = lane >> 4;
  const float* arow = E + (size_t)(tile * 16 + m) * DIN;

  CF acc[NT_N] = {};

#pragma unroll
  for (int ks = 0; ks < KSTEPS; ++ks) {
    int kA = ks * 32 + khalf * 8;
    f32x4 a0 = *(const f32x4*)(arow + kA);
    f32x4 a1 = *(const f32x4*)(arow + kA + 4);
    f32x4 a2 = *(const f32x4*)(arow + kA + 16);
    f32x4 a3 = *(const f32x4*)(arow + kA + 20);
    ABf afrag;
#pragma unroll
    for (int j = 0; j < 4; ++j) {
      afrag.e[j]      = (__bf16)a0[j];
      afrag.e[4 + j]  = (__bf16)a1[j];
      afrag.e[8 + j]  = (__bf16)a2[j];
      afrag.e[12 + j] = (__bf16)a3[j];
    }
#pragma unroll
    for (int n = 0; n < NT_N; ++n) {
      v16bf bfrag = Wf[(n * KSTEPS + ks) * 32 + lane];
      acc[n].v = __builtin_amdgcn_wmma_f32_16x16x32_bf16(
          false, afrag.v, false, bfrag, (short)0, acc[n].v, false, false);
    }
  }

  // C/D layout: lanes 0-15 -> rows v..v (M=v), lanes 16-31 -> M=8+v; N = lane%16
  int rbase = tile * 16 + khalf * 8;
  int cbase = lane & 15;
#pragma unroll
  for (int n = 0; n < NT_N; ++n) {
#pragma unroll
    for (int v = 0; v < 8; ++v) {
      P[(size_t)(rbase + v) * DOUT + n * 16 + cbase] = acc[n].e[v];
    }
  }
}

// ---------------------------------------------------------------------------
// Kernel 2: one wave32 per node; single-pass online masked softmax aggregation.
// Lane l owns dims {2l, 2l+1}; per-neighbor dot via 5-step shfl_xor reduction.
// P (51.2 MB) is fully L2-resident (192 MB L2) so gathers hit L2.
// ---------------------------------------------------------------------------
__device__ __forceinline__ float leaky(float x) { return x >= 0.f ? x : 0.2f * x; }

__device__ __forceinline__ float wave_sum(float v) {
#pragma unroll
  for (int off = 16; off > 0; off >>= 1) v += __shfl_xor(v, off, 32);
  return v;
}

__global__ void attn_kernel(const int* __restrict__ neighs,
                            const unsigned char* __restrict__ mask,
                            const int* __restrict__ dstidx,
                            const float* __restrict__ P,
                            const float* __restrict__ a_src,
                            const float* __restrict__ a_dst,
                            float* __restrict__ out, int B, int L) {
  int wave = threadIdx.x >> 5;
  int lane = threadIdx.x & 31;
  int b = blockIdx.x * 8 + wave;
  if (b >= B) return;                               // wave-uniform

  float2 as = ((const float2*)a_src)[lane];
  float2 ad = ((const float2*)a_dst)[lane];

  int d = dstidx[b];
  float2 df = ((const float2*)(P + (size_t)d * DOUT))[lane];
  float dattn = leaky(wave_sum(df.x * ad.x + df.y * ad.y));

  const int* nb = neighs + (size_t)b * L;
  const unsigned char* mk = mask + (size_t)b * L;

  float m = -1e30f, s = 0.f, ax = 0.f, ay = 0.f;
  for (int l = 0; l < L; ++l) {
    if (!mk[l]) continue;                           // wave-uniform skip
    if (l + 1 < L)
      __builtin_prefetch(P + (size_t)nb[l + 1] * DOUT, 0, 1);
    int idx = nb[l];
    float2 nf = ((const float2*)(P + (size_t)idx * DOUT))[lane];
    float sc = leaky(wave_sum(nf.x * as.x + nf.y * as.y)) + dattn;
    float mn = fmaxf(m, sc);
    float scale = __expf(m - mn);
    float w     = __expf(sc - mn);
    s  = s  * scale + w;
    ax = ax * scale + w * nf.x;
    ay = ay * scale + w * nf.y;
    m = mn;
  }
  float inv = 1.f / s;                              // mask[:,0]==True => s>0
  float2* o = (float2*)(out + (size_t)b * DOUT);
  o[lane] = make_float2(ax * inv, ay * inv);
}

// ---------------------------------------------------------------------------
extern "C" void kernel_launch(void* const* d_in, const int* in_sizes, int n_in,
                              void* d_out, int out_size, void* d_ws, size_t ws_size,
                              hipStream_t stream) {
  const int*           neighs = (const int*)d_in[0];
  const unsigned char* mask   = (const unsigned char*)d_in[1];  // jnp.bool_ = 1 byte
  const int*           dsti   = (const int*)d_in[2];
  const float*         embed  = (const float*)d_in[3];
  const float*         W      = (const float*)d_in[4];
  const float*         a_src  = (const float*)d_in[5];
  const float*         a_dst  = (const float*)d_in[6];
  float*               out    = (float*)d_out;

  const int B = in_sizes[2];            // 50000
  const int L = in_sizes[0] / B;        // 50
  const int N = in_sizes[3] / DIN;      // 200000

  // workspace: [0, 32KB) swizzled bf16 W fragments; [32KB, ...) P = embed' f32
  unsigned short* Wbf = (unsigned short*)d_ws;
  float* P = (float*)((char*)d_ws + 32768);

  prep_w_kernel<<<4, 256, 0, stream>>>(W, Wbf);

  int nTiles = N / 16;                  // 12500 (exact)
  proj_kernel<<<(nTiles + 7) / 8, 256, 0, stream>>>(embed, Wbf, P, nTiles);

  attn_kernel<<<(B + 7) / 8, 256, 0, stream>>>(neighs, mask, dsti, P,
                                               a_src, a_dst, out, B, L);
}
